// SeqLSTMLabelingModel_42107859370787
// MI455X (gfx1250) — matrix operand order repaired and checked by reading, not compiled, not run on hardware
//
#include <hip/hip_runtime.h>
#include <hip/hip_bf16.h>
#include <cstdint>
#include <cstddef>

// Problem dims
#define SDIM   512
#define SM1    511
#define M_TOT  65408      // 128 * 511
#define WDIM_  300
#define PDIM_  32
#define MDIM_  66
#define LDIM_  100
#define CONCAT 498
#define KPAD   512
#define UNITS  1024
#define GCOLS  4096
#define NCLS   43
#define NPAD   64

typedef __attribute__((ext_vector_type(16))) __bf16 v16bf;
typedef __attribute__((ext_vector_type(8)))  float  v8f;

union FragBF { v16bf v; uint4 q[2]; };

__device__ __forceinline__ unsigned short f2bf(float f) {
  union { float f; unsigned int u; } x; x.f = f;
  return (unsigned short)((x.u + 0x7FFFu + ((x.u >> 16) & 1u)) >> 16);
}
// Fast activations on v_exp_f32 / v_rcp_f32 (saturate correctly at +/-inf).
__device__ __forceinline__ float sigmoid_fast(float x) {
  return __builtin_amdgcn_rcpf(1.0f + __expf(-x));
}
__device__ __forceinline__ float tanh_fast(float x) {
  return 1.0f - 2.0f * __builtin_amdgcn_rcpf(1.0f + __expf(2.0f * x));
}
// One per-lane 16B async DMA: LDS[dstOff..+15] = MEM[src..+15], ASYNCcnt-tracked.
__device__ __forceinline__ void async_b128(unsigned dstOff, const void* src) {
  asm volatile("global_load_async_to_lds_b128 %0, %1, off"
               :: "v"(dstOff), "v"((unsigned long long)(uintptr_t)src)
               : "memory");
}
__device__ __forceinline__ void wait_async0() {
  asm volatile("s_wait_asynccnt 0x0" ::: "memory");
}

// ---------------------------------------------------------------------------
// Kernel 1: gather embeddings + concat + bf16 pack into xt[M_TOT][KPAD]
// row m = (b, s): b = m/511, s = m%511 + 1 (xt drops s=0; label uses s-1)
// ---------------------------------------------------------------------------
__global__ void pack_x_kernel(const int* __restrict__ words,
                              const int* __restrict__ pos,
                              const float* __restrict__ morph,
                              const int* __restrict__ labels,
                              const float* __restrict__ word_emb,
                              const float* __restrict__ pos_emb,
                              const float* __restrict__ label_emb,
                              unsigned short* __restrict__ xt) {
  const int m = blockIdx.x;
  const int b = m / SM1;
  const int s = m - b * SM1 + 1;
  const int wid = words[b * SDIM + s];
  const int pid = pos[b * SDIM + s];
  const int lid = labels[b * SDIM + s - 1];
  const float* wsrc = word_emb + (size_t)wid * WDIM_;
  const float* psrc = pos_emb + (size_t)pid * PDIM_;
  const float* msrc = morph + ((size_t)b * SDIM + s) * MDIM_;
  const float* lsrc = label_emb + (size_t)lid * LDIM_;
  unsigned short* dst = xt + (size_t)m * KPAD;
  for (int j = threadIdx.x; j < KPAD; j += blockDim.x) {
    float v;
    if (j < WDIM_)                      v = wsrc[j];
    else if (j < WDIM_ + PDIM_)         v = psrc[j - WDIM_];
    else if (j < WDIM_ + PDIM_ + MDIM_) v = msrc[j - (WDIM_ + PDIM_)];
    else if (j < CONCAT)                v = lsrc[j - (WDIM_ + PDIM_ + MDIM_)];
    else                                v = 0.0f;
    dst[j] = f2bf(v);
  }
}

// ---------------------------------------------------------------------------
// Kernel 2: Wx (498 x 4096 row-major) -> Wxp bf16 [col g][k], k padded to 512
// ---------------------------------------------------------------------------
__global__ void pack_wx_kernel(const float* __restrict__ Wx,
                               unsigned short* __restrict__ Wxp) {
  const size_t total = (size_t)GCOLS * KPAD;
  for (size_t i = (size_t)blockIdx.x * blockDim.x + threadIdx.x; i < total;
       i += (size_t)gridDim.x * blockDim.x) {
    int g = (int)(i >> 9);
    int k = (int)(i & (KPAD - 1));
    Wxp[i] = f2bf(k < CONCAT ? Wx[(size_t)k * GCOLS + g] : 0.0f);
  }
}

// ---------------------------------------------------------------------------
// Kernel 3: Wd (1024 x 43 row-major) -> Wdp bf16 [col c][k], c padded to 64
// ---------------------------------------------------------------------------
__global__ void pack_wd_kernel(const float* __restrict__ Wd,
                               unsigned short* __restrict__ Wdp) {
  const size_t total = (size_t)NPAD * UNITS;
  for (size_t i = (size_t)blockIdx.x * blockDim.x + threadIdx.x; i < total;
       i += (size_t)gridDim.x * blockDim.x) {
    int c = (int)(i >> 10);
    int k = (int)(i & (UNITS - 1));
    Wdp[i] = f2bf(c < NCLS ? Wd[(size_t)k * NCLS + c] : 0.0f);
  }
}

// ---------------------------------------------------------------------------
// Kernel 4: GEMM1 + activation, h = sig(zo)*tanh(sig(zi)*tanh(zg)) in bf16.
// Fragment-major LDS (16x32 frag = 1KB = 2 x 512B lane-major pieces):
//   A tile : 8 frag-rows x 16 ksteps x 1KB          = 131072 B
//   B bufs : 2 x (6 coltiles x 16 ksteps x 1KB)     = 196608 B  (async DMA)
// Total 327680 B = full 320KB WGP LDS.
// 8 waves = 4 M-waves x 2 U-waves; each wave: 2 M-subtiles x 3 gate-coltiles
// -> 6 WMMAs per k-step against only 10 ds_load_b128 (4 A + 6 B).
// ---------------------------------------------------------------------------
__global__ void gemm1_act_kernel(const unsigned short* __restrict__ xt,
                                 const unsigned short* __restrict__ Wxp,
                                 const float* __restrict__ bvec,
                                 unsigned short* __restrict__ hbuf) {
  extern __shared__ unsigned short smem1[];
  unsigned short* ldsA = smem1;                 // 65536 halfs
  const unsigned ldsBase = (unsigned)(uintptr_t)(void*)smem1;  // LDS byte offset
  const unsigned ldsBByte = ldsBase + 131072u;

  const int tid   = threadIdx.x;
  const int lane  = tid & 31;
  const int wave  = tid >> 5;
  const int mwave = wave >> 1;   // 0..3 : owns rows mwave*32 .. +31
  const int uwave = wave & 1;    // 0..1 : owns unit halves uwave*16 .. +15
  const int nrow  = lane & 15;
  const int half  = lane >> 4;
  const int mBase = blockIdx.x * 128;

  // Async-stage B chunk for unit base u0n into buffer bufSel (wave-split work).
  auto stageB_async = [&](int u0n, unsigned bufSel) {
#pragma unroll
    for (int i = 0; i < 12; ++i) {
      const int f   = i * 8 + wave;       // fragment id: ct = f>>4, ks = f&15
      const int ct  = f >> 4;
      const int ksf = f & 15;
      const int col = ct * 16 + nrow;     // 0..95
      const int gate = col >> 5;
      const int goff = (gate == 0) ? 0 : ((gate == 1) ? 2048 : 3072);
      const int g = goff + u0n + (col & 31);
      const unsigned short* src = Wxp + (size_t)g * KPAD + ksf * 32 + half * 16;
      const unsigned dst = ldsBByte + bufSel * 98304u +
                           (unsigned)f * 1024u + (unsigned)lane * 16u;
      async_b128(dst, src);            // piece0: halfs 0..7 of lane chunk
      async_b128(dst + 512u, src + 8); // piece1: halfs 8..15
    }
  };

  // Kick off B chunk 0 DMA, then async-stage the A tile (no VGPR round-trip).
  stageB_async(0, 0u);
  for (int c = tid; c < 8192; c += 256) {  // 8192 x 16B chunks
    const int frag = c >> 6;               // fragRow = frag>>4, ks = frag&15
    const int cc = c & 63;
    const int piece = cc >> 5;
    const int ln = cc & 31;
    const int row = (frag >> 4) * 16 + (ln & 15);
    const int k = (frag & 15) * 32 + (ln >> 4) * 16 + piece * 8;
    async_b128(ldsBase + (unsigned)c * 16u,
               xt + (size_t)(mBase + row) * KPAD + k);
  }
  wait_async0();
  __syncthreads();

  for (int uc = 0; uc < UNITS / 32; ++uc) {
    const int u0 = uc * 32;
    const unsigned cur = (unsigned)(uc & 1);
    if (uc + 1 < UNITS / 32) stageB_async(u0 + 32, cur ^ 1u);  // overlap DMA

    const unsigned short* bufB = smem1 + 65536 + cur * 49152u;

    // A fragment rows for this wave: mwave*2 + msub
    auto loadA = [&](int ks, FragBF* a) {
#pragma unroll
      for (int msub = 0; msub < 2; ++msub) {
        const unsigned short* p =
            ldsA + ((mwave * 2 + msub) * 16 + ks) * 512 + lane * 8;
        a[msub].q[0] = *(const uint4*)(p);
        a[msub].q[1] = *(const uint4*)(p + 256);
      }
    };
    // B col-tiles for this wave: gate*2 + uwave, gate = 0..2
    auto loadB = [&](int ks, FragBF* bb) {
#pragma unroll
      for (int gate = 0; gate < 3; ++gate) {
        const unsigned short* p =
            bufB + ((gate * 2 + uwave) * 16 + ks) * 512 + lane * 8;
        bb[gate].q[0] = *(const uint4*)(p);
        bb[gate].q[1] = *(const uint4*)(p + 256);
      }
    };

    v8f acc[6] = {};  // [msub*3 + gate]
    FragBF aF[2][2];
    FragBF bF[2][3];
    loadA(0, aF[0]);
    loadB(0, bF[0]);
#pragma unroll
    for (int ks = 0; ks < KPAD / 32; ++ks) {
      const int pcur = ks & 1;
      if (ks + 1 < KPAD / 32) {   // prefetch next k-step while WMMAs run
        loadA(ks + 1, aF[pcur ^ 1]);
        loadB(ks + 1, bF[pcur ^ 1]);
      }
#pragma unroll
      for (int msub = 0; msub < 2; ++msub) {
#pragma unroll
        for (int gate = 0; gate < 3; ++gate) {
          acc[msub * 3 + gate] = __builtin_amdgcn_wmma_f32_16x16x32_bf16(
              false, aF[pcur][msub].v, false, bF[pcur][gate].v, (short)0,
              acc[msub * 3 + gate], false, false);
        }
      }
    }

    // Epilogue: bias + fast activations, write h bf16
    {
      const int u = u0 + uwave * 16 + nrow;
      const float bi = bvec[u];
      const float bg = bvec[2048 + u];
      const float bo = bvec[3072 + u];
#pragma unroll
      for (int msub = 0; msub < 2; ++msub) {
#pragma unroll
        for (int r = 0; r < 8; ++r) {
          const int mrow = mBase + mwave * 32 + msub * 16 + r + half * 8;
          const float zi = acc[msub * 3 + 0][r] + bi;
          const float zg = acc[msub * 3 + 1][r] + bg;
          const float zo = acc[msub * 3 + 2][r] + bo;
          const float cc = sigmoid_fast(zi) * tanh_fast(zg);
          const float hh = sigmoid_fast(zo) * tanh_fast(cc);
          hbuf[(size_t)mrow * UNITS + u] = f2bf(hh);
        }
      }
    }
    wait_async0();     // this wave's DMA for next buffer done
    __syncthreads();   // all waves done reading cur / writing nxt
  }
}

// ---------------------------------------------------------------------------
// Kernel 5: GEMM2  scores = h @ Wd + bd (M=65408, K=1024, N=43 pad 64), f32 out
// Fragment-major Wd tile: 4 coltiles x 32 ksteps x 1KB = 131072 B, async fill.
// ---------------------------------------------------------------------------
__global__ void gemm2_kernel(const unsigned short* __restrict__ hbuf,
                             const unsigned short* __restrict__ Wdp,
                             const float* __restrict__ bd,
                             float* __restrict__ out) {
  extern __shared__ unsigned short smem2[];
  unsigned short* ldsB = smem2;  // 65536 halfs
  const unsigned ldsBase = (unsigned)(uintptr_t)(void*)smem2;

  const int tid  = threadIdx.x;
  const int lane = tid & 31;
  const int wave = tid >> 5;
  const int nrow = lane & 15;
  const int half = lane >> 4;
  const int mBase = blockIdx.x * 128;

  // Async-stage Wd tile: 8192 x 16B chunks, fragment-major
  for (int c = tid; c < 8192; c += 256) {
    const int frag = c >> 6;               // ct = frag>>5, ks = frag&31
    const int cc = c & 63;
    const int piece = cc >> 5;
    const int ln = cc & 31;
    const int col = (frag >> 5) * 16 + (ln & 15);
    const int k = (frag & 31) * 32 + (ln >> 4) * 8 + piece * 16;
    async_b128(ldsBase + (unsigned)c * 16u, Wdp + (size_t)col * UNITS + k);
  }
  wait_async0();
  __syncthreads();

  const unsigned short* arow = hbuf + (size_t)(mBase + wave * 16 + nrow) * UNITS;
  auto loadA = [&](int ks, FragBF& a) {   // from global h (row-major)
    const uint4* pa = (const uint4*)(arow + ks * 32 + half * 8);
    a.q[0] = pa[0];
    a.q[1] = pa[2];
  };
  auto loadB = [&](int ks, FragBF* bb) {
#pragma unroll
    for (int j = 0; j < 4; ++j) {
      const unsigned short* p = ldsB + (j * 32 + ks) * 512 + lane * 8;
      bb[j].q[0] = *(const uint4*)(p);
      bb[j].q[1] = *(const uint4*)(p + 256);
    }
  };

  v8f acc[4] = {};
  FragBF aF[2];
  FragBF bF[2][4];
  loadA(0, aF[0]);
  loadB(0, bF[0]);
#pragma unroll
  for (int ks = 0; ks < UNITS / 32; ++ks) {
    const int pcur = ks & 1;
    if (ks + 1 < UNITS / 32) {
      loadA(ks + 1, aF[pcur ^ 1]);
      loadB(ks + 1, bF[pcur ^ 1]);
    }
#pragma unroll
    for (int j = 0; j < 4; ++j) {
      acc[j] = __builtin_amdgcn_wmma_f32_16x16x32_bf16(
          false, aF[pcur].v, false, bF[pcur][j].v, (short)0, acc[j], false, false);
    }
  }

#pragma unroll
  for (int nsub = 0; nsub < 4; ++nsub) {
    const int c = nsub * 16 + nrow;
    if (c < NCLS) {
      const float bdv = bd[c];
#pragma unroll
      for (int r = 0; r < 8; ++r) {
        const int mrow = mBase + wave * 16 + r + half * 8;
        out[(size_t)mrow * NCLS + c] = acc[nsub][r] + bdv;
      }
    }
  }
}

// ---------------------------------------------------------------------------
extern "C" void kernel_launch(void* const* d_in, const int* in_sizes, int n_in,
                              void* d_out, int out_size, void* d_ws, size_t ws_size,
                              hipStream_t stream) {
  const int*   words     = (const int*)d_in[0];
  const int*   pos       = (const int*)d_in[1];
  const float* morph     = (const float*)d_in[2];
  const int*   labels    = (const int*)d_in[3];
  const float* word_emb  = (const float*)d_in[4];
  const float* pos_emb   = (const float*)d_in[5];
  const float* label_emb = (const float*)d_in[6];
  const float* Wx        = (const float*)d_in[7];
  // d_in[8] = Wh : dead in the reference computation
  const float* bvec      = (const float*)d_in[9];
  const float* Wd        = (const float*)d_in[10];
  const float* bd        = (const float*)d_in[11];
  float* out = (float*)d_out;

  // Workspace carve-up (bf16 halves), all 16B-aligned offsets
  unsigned short* xt   = (unsigned short*)d_ws;                 // M_TOT*512
  unsigned short* Wxp  = xt  + (size_t)M_TOT * KPAD;            // 4096*512
  unsigned short* Wdp  = Wxp + (size_t)GCOLS * KPAD;            // 64*1024
  unsigned short* hbuf = Wdp + (size_t)NPAD * UNITS;            // M_TOT*1024

  pack_x_kernel<<<M_TOT, 128, 0, stream>>>(words, pos, morph, labels, word_emb,
                                           pos_emb, label_emb, xt);
  pack_wx_kernel<<<512, 256, 0, stream>>>(Wx, Wxp);
  pack_wd_kernel<<<64, 256, 0, stream>>>(Wd, Wdp);

  const size_t lds1 = 131072 + 2 * 98304;   // 327680 B = 320 KB WGP LDS
  gemm1_act_kernel<<<M_TOT / 128, 256, lds1, stream>>>(xt, Wxp, bvec, hbuf);

  const size_t lds2 = 131072;
  gemm2_kernel<<<M_TOT / 128, 256, lds2, stream>>>(hbuf, Wdp, bd, out);
}
